// Optim1_72224170049943
// MI455X (gfx1250) — matrix operand order repaired and checked by reading, not compiled, not run on hardware
//
#include <hip/hip_runtime.h>

// ---------------- problem constants ----------------
#define Bb 64
#define Td 2048
#define Xd 128
#define Yd 256
#define Ud 64
#define Zd 32
#define Mrows (Bb * Td)   // 131072 rows of the time-parallel GEMMs
#define Lc 128            // scan chunk length
#define NCH (Td / Lc)     // 16 chunks

// ---------------- WMMA types ----------------
typedef __attribute__((ext_vector_type(16))) __bf16 bf16x16;
typedef __attribute__((ext_vector_type(8)))  float  v8f;

union Frag16 { bf16x16 v; unsigned u[8]; };

__device__ __forceinline__ unsigned pack2bf(float a, float b) {
  union { __bf16 h[2]; unsigned u; } p;
  p.h[0] = (__bf16)a; p.h[1] = (__bf16)b;
  return p.u;
}

__device__ __forceinline__ unsigned load_pack2(const float* p) {
  float2 t = *(const float2*)p;          // global_load_b64
  return pack2bf(t.x, t.y);
}

// =====================================================================
// Kernel 1: drive = y @ Kw^T + u @ Bw^T + Kb + Bb      [Mrows x Xd] f32
// block: 256 thr (8 waves), computes a 128(M) x 128(N) tile.
// =====================================================================
__global__ __launch_bounds__(256)
void drive_gemm_kernel(const float* __restrict__ y, const float* __restrict__ u,
                       const float* __restrict__ Kw, const float* __restrict__ Kb,
                       const float* __restrict__ Bw, const float* __restrict__ Bbias,
                       float* __restrict__ drive)
{
  __shared__ __bf16 aT[128 * 32];   // activation tile [m][k], k-contiguous
  __shared__ __bf16 bT[128 * 32];   // weight tile     [n][k], k-contiguous
  unsigned* aT32 = (unsigned*)aT;
  unsigned* bT32 = (unsigned*)bT;

  const int tid  = threadIdx.x;
  const int wave = tid >> 5, lane = tid & 31;
  const int half = lane >> 4, l16 = lane & 15;
  const long r0  = (long)blockIdx.x * 128;

  v8f acc[8];
#pragma unroll
  for (int i = 0; i < 8; ++i) acc[i] = {};

  for (int phase = 0; phase < 2; ++phase) {
    const float* src = phase ? u  : y;
    const float* W   = phase ? Bw : Kw;
    const int Kdim   = phase ? Ud : Yd;
    for (int kb = 0; kb < Kdim; kb += 32) {
      __syncthreads();
      // stage 128x32 tiles as packed-bf16 dwords (2048 dwords each, 8/thread)
#pragma unroll
      for (int i = 0; i < 8; ++i) {
        int e = i * 256 + tid;          // dword idx: row = e/16, pair = e%16
        int row = e >> 4, pr = e & 15;
        int k = kb + pr * 2;
        aT32[e] = load_pack2(src + (r0 + row) * Kdim + k);
        bT32[e] = load_pack2(W + (long)row * Kdim + k);
      }
      __syncthreads();

      Frag16 af;
      const int m = wave * 16 + l16;
#pragma unroll
      for (int v = 0; v < 8; ++v) {
        int k0 = half * 16 + 2 * v;
        af.u[v] = aT32[m * 16 + (k0 >> 1)];
      }
#pragma unroll
      for (int nt = 0; nt < 8; ++nt) {
        Frag16 bfr;
        const int n = nt * 16 + l16;
#pragma unroll
        for (int v = 0; v < 8; ++v)
          bfr.u[v] = bT32[n * 16 + half * 8 + v];
        acc[nt] = __builtin_amdgcn_wmma_f32_16x16x32_bf16(
            false, af.v, false, bfr.v, (short)0, acc[nt], false, false);
      }
    }
  }

#pragma unroll
  for (int nt = 0; nt < 8; ++nt) {
    const int n = nt * 16 + l16;
    const float bias = Kb[n] + Bbias[n];
#pragma unroll
    for (int r = 0; r < 8; ++r) {
      const int m = wave * 16 + half * 8 + r;
      drive[(r0 + m) * Xd + n] = acc[nt][r] + bias;
    }
  }
}

// =====================================================================
// Kernel 2: ApowL = A^Lc (bf16, row-major [out][in]) via 7 WMMA squarings.
// Single workgroup; Pbuf = 2 x 128x128 f32 ping-pong in ws.
// Dynamic LDS: rowL[m][k] + colL[n][k] bf16 images (64 KB).
// =====================================================================
__global__ __launch_bounds__(256)
void apow_kernel(const float* __restrict__ Aw, float* __restrict__ Pbuf,
                 __bf16* __restrict__ ApowL)
{
  extern __shared__ __bf16 smem[];
  __bf16* rowL = smem;              // rowL[m*128+k] = P[m][k]
  __bf16* colL = smem + 128 * 128;  // colL[n*128+k] = P[k][n]
  unsigned* row32 = (unsigned*)rowL;
  unsigned* col32 = (unsigned*)colL;

  const int tid = threadIdx.x, wave = tid >> 5, lane = tid & 31;
  const int half = lane >> 4, l16 = lane & 15;

  for (int e = tid; e < 128 * 128; e += 256) Pbuf[e] = Aw[e];
  __threadfence_block();
  __syncthreads();

  for (int it = 0; it < 7; ++it) {
    const float* P = Pbuf + (it & 1) * 16384;
    float*       Q = Pbuf + ((it + 1) & 1) * 16384;
    for (int e = tid; e < 128 * 128; e += 256) {
      int r = e >> 7, c = e & 127;
      float v = P[e];
      rowL[e] = (__bf16)v;
      colL[c * 128 + r] = (__bf16)v;
    }
    __syncthreads();

    const int mt = wave;
    v8f acc[8];
#pragma unroll
    for (int i = 0; i < 8; ++i) acc[i] = {};
    for (int ks = 0; ks < 4; ++ks) {
      Frag16 af;
      const int m = mt * 16 + l16;
#pragma unroll
      for (int v = 0; v < 8; ++v) {
        int k0 = ks * 32 + half * 16 + 2 * v;
        af.u[v] = row32[m * 64 + (k0 >> 1)];
      }
#pragma unroll
      for (int nt = 0; nt < 8; ++nt) {
        Frag16 bfr;
        const int n = nt * 16 + l16;
#pragma unroll
        for (int v = 0; v < 8; ++v) {
          int k = ks * 32 + half * 16 + 2 * v;
          bfr.u[v] = col32[n * 64 + (k >> 1)];
        }
        acc[nt] = __builtin_amdgcn_wmma_f32_16x16x32_bf16(
            false, af.v, false, bfr.v, (short)0, acc[nt], false, false);
      }
    }
#pragma unroll
    for (int nt = 0; nt < 8; ++nt) {
      const int n = nt * 16 + l16;
#pragma unroll
      for (int r = 0; r < 8; ++r) {
        const int m = mt * 16 + half * 8 + r;
        Q[m * 128 + n] = acc[nt][r];
      }
    }
    __threadfence_block();
    __syncthreads();
  }

  const float* PF = Pbuf + 16384;  // after 7 iters result is in buffer 1
  for (int e = tid; e < 128 * 128; e += 256) ApowL[e] = (__bf16)PF[e];
}

// =====================================================================
// Kernel 3: within-chunk recurrence  x <- A x + A_b + d_t.
// grid = 4 batch-tiles x 16 chunks; 8 waves, wave w owns output cols
// [16w,16w+16). State carried f32 in accumulators; bf16 multiplicand
// is exchanged through LDS each step. WMMA C operand = A_b + drive_t.
// passC=0: start from 0, emit final x as v_c.  passC=1: start from
// x_start, emit pre-update states to d_out.
// =====================================================================
__global__ __launch_bounds__(256)
void chunk_scan_kernel(const float* __restrict__ drive, const float* __restrict__ Aw,
                       const float* __restrict__ Ab, const float* __restrict__ xstart,
                       float* __restrict__ vout, float* __restrict__ states,
                       int passC)
{
  __shared__ __bf16 colA[128 * 128];  // colA[n*128+k] = Aw[n][k] (== Aw row-major)
  __shared__ __bf16 xL[16 * 128];     // xL[m*128+k] bf16 state tile
  unsigned* colA32 = (unsigned*)colA;
  unsigned* xL32   = (unsigned*)xL;

  const int tid = threadIdx.x, wave = tid >> 5, lane = tid & 31;
  const int half = lane >> 4, l16 = lane & 15;
  const int mt = blockIdx.x & 3;       // batch tile (64/16)
  const int c  = blockIdx.x >> 2;      // chunk index
  const int b0 = mt * 16;
  const long t0 = (long)c * Lc;

  for (int e = tid; e < 128 * 64; e += 256) {   // 8192 dwords
    int n = e >> 6, pi = e & 63;
    colA32[e] = load_pack2(Aw + n * 128 + pi * 2);
  }
  __syncthreads();

  // hoist the 4 K-step B-fragments of this wave's N-tile (A is constant)
  const int nt = wave;
  Frag16 bfr[4];
#pragma unroll
  for (int ks = 0; ks < 4; ++ks) {
    const int n = nt * 16 + l16;
#pragma unroll
    for (int v = 0; v < 8; ++v) {
      int k = ks * 32 + half * 16 + 2 * v;
      bfr[ks].u[v] = colA32[n * 64 + (k >> 1)];
    }
  }

  const int ncol = nt * 16 + l16;
  const float abias = Ab[ncol];

  v8f acc = {};
  if (passC) {
#pragma unroll
    for (int r = 0; r < 8; ++r) {
      const int m = half * 8 + r;
      acc[r] = xstart[((long)(b0 + m) * NCH + c) * Xd + ncol];
    }
  }

  for (int j = 0; j < Lc; ++j) {
    const long t = t0 + j;
    // emit pre-update state (pass C) and publish bf16 multiplicand
#pragma unroll
    for (int r = 0; r < 8; ++r) {
      const int m = half * 8 + r;
      if (passC)
        states[((long)(b0 + m) * Td + t) * Xd + ncol] = acc[r];
      xL[m * 128 + ncol] = (__bf16)acc[r];
    }
    __syncthreads();

    v8f nacc;
#pragma unroll
    for (int r = 0; r < 8; ++r) {
      const int m = half * 8 + r;
      nacc[r] = abias + drive[((long)(b0 + m) * Td + t) * Xd + ncol];
    }
#pragma unroll
    for (int ks = 0; ks < 4; ++ks) {
      Frag16 af;
      const int m = l16;
#pragma unroll
      for (int v = 0; v < 8; ++v) {
        int k0 = ks * 32 + half * 16 + 2 * v;
        af.u[v] = xL32[m * 64 + (k0 >> 1)];
      }
      nacc = __builtin_amdgcn_wmma_f32_16x16x32_bf16(
          false, af.v, false, bfr[ks].v, (short)0, nacc, false, false);
    }
    acc = nacc;
    __syncthreads();
  }

  if (!passC) {
#pragma unroll
    for (int r = 0; r < 8; ++r) {
      const int m = half * 8 + r;
      vout[((long)(b0 + m) * NCH + c) * Xd + ncol] = acc[r];
    }
  }
}

// =====================================================================
// Kernel 4: chain chunk boundaries:  x_start[c+1] = A^L x_start[c] + v_c.
// Single workgroup, 16 sequential chunk steps, WMMA with v_c as C.
// wave w: batch tile mt = w&3, N-tiles [(w>>2)*4, +4).
// =====================================================================
__global__ __launch_bounds__(256)
void combine_kernel(const __bf16* __restrict__ ApowL, const float* __restrict__ vbuf,
                    float* __restrict__ xstart)
{
  __shared__ __bf16 colP[128 * 128];  // colP[n*128+k] = A^L[n][k]
  __shared__ __bf16 xL[64 * 128];     // bf16 state, all 64 batches
  unsigned* colP32 = (unsigned*)colP;
  unsigned* xL32   = (unsigned*)xL;

  const int tid = threadIdx.x, wave = tid >> 5, lane = tid & 31;
  const int half = lane >> 4, l16 = lane & 15;

  const unsigned* Ap32 = (const unsigned*)ApowL;   // already bf16-packed
  for (int e = tid; e < 128 * 64; e += 256) colP32[e] = Ap32[e];

  const int mt = wave & 3;
  const int ntbase = (wave >> 2) * 4;
  const int b0 = mt * 16;

  v8f acc[4];
#pragma unroll
  for (int q = 0; q < 4; ++q) acc[q] = {};
  __syncthreads();

  for (int c = 0; c < NCH; ++c) {
#pragma unroll
    for (int q = 0; q < 4; ++q) {
      const int n = (ntbase + q) * 16 + l16;
#pragma unroll
      for (int r = 0; r < 8; ++r) {
        const int m = b0 + half * 8 + r;
        xstart[((long)m * NCH + c) * Xd + n] = acc[q][r];
        xL[m * 128 + n] = (__bf16)acc[q][r];
      }
    }
    __syncthreads();

    v8f nacc[4];
#pragma unroll
    for (int q = 0; q < 4; ++q) {
      const int n = (ntbase + q) * 16 + l16;
#pragma unroll
      for (int r = 0; r < 8; ++r) {
        const int m = b0 + half * 8 + r;
        nacc[q][r] = vbuf[((long)m * NCH + c) * Xd + n];
      }
    }
#pragma unroll
    for (int ks = 0; ks < 4; ++ks) {
      Frag16 af;
      const int m = mt * 16 + l16;
#pragma unroll
      for (int v = 0; v < 8; ++v) {
        int k0 = ks * 32 + half * 16 + 2 * v;
        af.u[v] = xL32[m * 64 + (k0 >> 1)];
      }
#pragma unroll
      for (int q = 0; q < 4; ++q) {
        Frag16 bfr;
        const int n = (ntbase + q) * 16 + l16;
#pragma unroll
        for (int v = 0; v < 8; ++v) {
          int k = ks * 32 + half * 16 + 2 * v;
          bfr.u[v] = colP32[n * 64 + (k >> 1)];
        }
        nacc[q] = __builtin_amdgcn_wmma_f32_16x16x32_bf16(
            false, af.v, false, bfr.v, (short)0, nacc[q], false, false);
      }
    }
#pragma unroll
    for (int q = 0; q < 4; ++q) acc[q] = nacc[q];
    __syncthreads();
  }
}

// =====================================================================
// Kernel 5: z = states @ Czw^T + Czb   [Mrows x 32]
// =====================================================================
__global__ __launch_bounds__(256)
void z_gemm_kernel(const float* __restrict__ states, const float* __restrict__ Czw,
                   const float* __restrict__ Czb, float* __restrict__ z)
{
  __shared__ __bf16 aT[128 * 32];   // states tile [m][k]
  __shared__ __bf16 cz[32 * 128];   // cz[n*128+k] = Czw[n][k], staged once
  unsigned* aT32 = (unsigned*)aT;
  unsigned* cz32 = (unsigned*)cz;

  const int tid = threadIdx.x, wave = tid >> 5, lane = tid & 31;
  const int half = lane >> 4, l16 = lane & 15;
  const long r0 = (long)blockIdx.x * 128;

  for (int e = tid; e < 32 * 64; e += 256) {
    int n = e >> 6, pi = e & 63;
    cz32[e] = load_pack2(Czw + n * 128 + pi * 2);
  }

  v8f acc[2];
  acc[0] = {}; acc[1] = {};
  for (int kb = 0; kb < 128; kb += 32) {
    __syncthreads();
#pragma unroll
    for (int i = 0; i < 8; ++i) {
      int e = i * 256 + tid;
      int row = e >> 4, pr = e & 15;
      aT32[e] = load_pack2(states + (r0 + row) * Xd + kb + pr * 2);
    }
    __syncthreads();

    Frag16 af;
    const int m = wave * 16 + l16;
#pragma unroll
    for (int v = 0; v < 8; ++v) {
      int k0 = half * 16 + 2 * v;
      af.u[v] = aT32[m * 16 + (k0 >> 1)];
    }
#pragma unroll
    for (int nt = 0; nt < 2; ++nt) {
      Frag16 bfr;
      const int n = nt * 16 + l16;
#pragma unroll
      for (int v = 0; v < 8; ++v) {
        int k = kb + half * 16 + 2 * v;
        bfr.u[v] = cz32[n * 64 + (k >> 1)];
      }
      acc[nt] = __builtin_amdgcn_wmma_f32_16x16x32_bf16(
          false, af.v, false, bfr.v, (short)0, acc[nt], false, false);
    }
  }
#pragma unroll
  for (int nt = 0; nt < 2; ++nt) {
    const int n = nt * 16 + l16;
    const float bias = Czb[n];
#pragma unroll
    for (int r = 0; r < 8; ++r) {
      const int m = wave * 16 + half * 8 + r;
      z[(r0 + m) * Zd + n] = acc[nt][r] + bias;
    }
  }
}

// =====================================================================
extern "C" void kernel_launch(void* const* d_in, const int* in_sizes, int n_in,
                              void* d_out, int out_size, void* d_ws, size_t ws_size,
                              hipStream_t stream)
{
  (void)in_sizes; (void)n_in; (void)out_size; (void)ws_size;

  const float* y     = (const float*)d_in[0];
  const float* u     = (const float*)d_in[1];
  const float* Aw    = (const float*)d_in[2];
  const float* Ab    = (const float*)d_in[3];
  const float* Kw    = (const float*)d_in[4];
  const float* Kb    = (const float*)d_in[5];
  const float* Bw    = (const float*)d_in[6];
  const float* Bbias = (const float*)d_in[7];
  const float* Czw   = (const float*)d_in[8];
  const float* Czb   = (const float*)d_in[9];

  float* z      = (float*)d_out;                         // [B,T,Z]
  float* states = (float*)d_out + (size_t)Bb * Td * Zd;  // [B,T,X]

  char* ws = (char*)d_ws;
  float* drive  = (float*)ws;  ws += (size_t)Mrows * Xd * sizeof(float);   // 64 MB
  float* vbuf   = (float*)ws;  ws += (size_t)Bb * NCH * Xd * sizeof(float);
  float* xstart = (float*)ws;  ws += (size_t)Bb * NCH * Xd * sizeof(float);
  float* Pbuf   = (float*)ws;  ws += (size_t)2 * 128 * 128 * sizeof(float);
  __bf16* ApowL = (__bf16*)ws; ws += (size_t)128 * 128 * sizeof(__bf16);

  drive_gemm_kernel<<<Mrows / 128, 256, 0, stream>>>(y, u, Kw, Kb, Bw, Bbias, drive);
  apow_kernel<<<1, 256, 64 * 1024, stream>>>(Aw, Pbuf, ApowL);
  chunk_scan_kernel<<<4 * NCH, 256, 0, stream>>>(drive, Aw, Ab, xstart, vbuf, states, 0);
  combine_kernel<<<1, 256, 0, stream>>>(ApowL, vbuf, xstart);
  chunk_scan_kernel<<<4 * NCH, 256, 0, stream>>>(drive, Aw, Ab, xstart, vbuf, states, 1);
  z_gemm_kernel<<<Mrows / 128, 256, 0, stream>>>(states, Czw, Czb, z);
}
